// Classifier3Stage_38749194944898
// MI455X (gfx1250) — matrix operand order repaired and checked by reading, not compiled, not run on hardware
//
#include <hip/hip_runtime.h>
#include <hip/hip_bf16.h>
#include <math.h>

#define HH   448
#define WW   512
#define NPIX (HH * WW)
#define NC1  16
#define NC2  12
#define NC3  8
#define PP2  2
#define PP3  4

typedef __attribute__((ext_vector_type(16))) __bf16 v16bf;
typedef __attribute__((ext_vector_type(8)))  float  v8f;

__device__ __forceinline__ float lrelu(float x) { return x > 0.f ? x : 0.01f * x; }

__device__ __forceinline__ float bcast_lane(float v, int i) {
    return __int_as_float(__builtin_amdgcn_readlane(__float_as_int(v), i));
}

// ---------------------------------------------------------------------------
// Stage 1: per-scanline 3-layer MLP via WMMA bf16. One block per row h.
// 8 waves/block, each wave owns 16-pixel tiles (M=16, K=32 per layer).
// ---------------------------------------------------------------------------
__global__ __launch_bounds__(256) void stage1_kernel(
    const float* __restrict__ x_in,
    const float* __restrict__ w1_0, const float* __restrict__ b1_0,
    const float* __restrict__ w1_1, const float* __restrict__ b1_1,
    const float* __restrict__ w1_2, const float* __restrict__ b1_2,
    float* __restrict__ out, int* __restrict__ inds1_ws)
{
    __shared__ __bf16 wl0[32 * 32];   // [out][in]
    __shared__ __bf16 wl1[32 * 32];
    __shared__ __bf16 wl2[16 * 32];
    __shared__ float  bl0[32], bl1[32], bl2[16];
    __shared__ __bf16 act[8][16][32];   // per-wave [pixel][channel]
    __shared__ float  logit[8][16][16]; // per-wave [pixel][class]

    const int h   = blockIdx.x;
    const int tid = threadIdx.x;

    for (int i = tid; i < 1024; i += 256) wl0[i] = (__bf16)w1_0[(size_t)h * 1024 + i];
    for (int i = tid; i < 1024; i += 256) wl1[i] = (__bf16)w1_1[(size_t)h * 1024 + i];
    for (int i = tid; i < 512;  i += 256) wl2[i] = (__bf16)w1_2[(size_t)h * 512  + i];
    if (tid < 32) { bl0[tid] = b1_0[h * 32 + tid]; bl1[tid] = b1_1[h * 32 + tid]; }
    if (tid < 16) { bl2[tid] = b1_2[h * 16 + tid]; }
    __syncthreads();

    const int wave = tid >> 5;
    const int lane = tid & 31;
    const int half = lane >> 4;   // 0 or 1
    const int lm   = lane & 15;   // pixel (A/C rows) or column (B/C cols)

    for (int tile = wave; tile < WW / 16; tile += 8) {
        const int w0 = tile * 16;

        // ---- A fragment from x_in channels [0,32): A(m=lm, k per ISA layout)
        v16bf a;
        #pragma unroll
        for (int e = 0; e < 16; ++e) {
            int k = (e & 7) + 8 * half + 16 * (e >> 3);
            a[e] = (__bf16)x_in[(size_t)k * NPIX + (size_t)h * WW + w0 + lm];
        }

        // ---- layer 0: 32 -> 32 (two 16-wide N tiles)
        {
            v16bf b0, b1v;
            #pragma unroll
            for (int e = 0; e < 16; ++e) {
                b0[e]  = wl0[lm * 32        + (e & 7) + 8 * half + 16 * (e >> 3)]; // B(k=(e&7)+8*half+16*(e>>3), n=lm)
                b1v[e] = wl0[(lm + 16) * 32 + (e & 7) + 8 * half + 16 * (e >> 3)]; // n = lm+16
            }
            v8f z{};
            v8f d0 = __builtin_amdgcn_wmma_f32_16x16x32_bf16(false, a, false, b0,  (short)0, z, false, false);
            v8f d1 = __builtin_amdgcn_wmma_f32_16x16x32_bf16(false, a, false, b1v, (short)0, z, false, false);
            #pragma unroll
            for (int e = 0; e < 8; ++e) {            // D(m=e+8*half, n=lm)
                int m = e + 8 * half;
                act[wave][m][lm]      = (__bf16)lrelu(d0[e] + bl0[lm]);
                act[wave][m][lm + 16] = (__bf16)lrelu(d1[e] + bl0[lm + 16]);
            }
            asm volatile("s_wait_dscnt 0" ::: "memory");
        }

        // ---- layer 1: 32 -> 32
        #pragma unroll
        for (int e = 0; e < 16; ++e) {
            int k = (e & 7) + 8 * half + 16 * (e >> 3);
            a[e] = act[wave][lm][k];
        }
        asm volatile("s_wait_dscnt 0" ::: "memory");
        {
            v16bf b0, b1v;
            #pragma unroll
            for (int e = 0; e < 16; ++e) {
                b0[e]  = wl1[lm * 32        + (e & 7) + 8 * half + 16 * (e >> 3)];
                b1v[e] = wl1[(lm + 16) * 32 + (e & 7) + 8 * half + 16 * (e >> 3)];
            }
            v8f z{};
            v8f d0 = __builtin_amdgcn_wmma_f32_16x16x32_bf16(false, a, false, b0,  (short)0, z, false, false);
            v8f d1 = __builtin_amdgcn_wmma_f32_16x16x32_bf16(false, a, false, b1v, (short)0, z, false, false);
            #pragma unroll
            for (int e = 0; e < 8; ++e) {
                int m = e + 8 * half;
                act[wave][m][lm]      = (__bf16)lrelu(d0[e] + bl1[lm]);
                act[wave][m][lm + 16] = (__bf16)lrelu(d1[e] + bl1[lm + 16]);
            }
            asm volatile("s_wait_dscnt 0" ::: "memory");
        }

        // ---- layer 2: 32 -> 16 logits
        #pragma unroll
        for (int e = 0; e < 16; ++e) {
            int k = (e & 7) + 8 * half + 16 * (e >> 3);
            a[e] = act[wave][lm][k];
        }
        asm volatile("s_wait_dscnt 0" ::: "memory");
        {
            v16bf b2;
            #pragma unroll
            for (int e = 0; e < 16; ++e)
                b2[e] = wl2[lm * 32 + (e & 7) + 8 * half + 16 * (e >> 3)];
            v8f z{};
            v8f dl = __builtin_amdgcn_wmma_f32_16x16x32_bf16(false, a, false, b2, (short)0, z, false, false);
            #pragma unroll
            for (int e = 0; e < 8; ++e)
                logit[wave][e + 8 * half][lm] = dl[e] + bl2[lm];
            asm volatile("s_wait_dscnt 0" ::: "memory");
        }

        // ---- per-pixel argmax + entropy (lanes 0..15, one pixel each)
        if (half == 0) {
            float vmax = -3.4e38f; int imax = 0;
            #pragma unroll
            for (int c = 0; c < 16; ++c) {
                float v = logit[wave][lm][c];
                if (v > vmax) { vmax = v; imax = c; }
            }
            float s1 = 0.f, s2 = 0.f;
            #pragma unroll
            for (int c = 0; c < 16; ++c) {
                float v = logit[wave][lm][c];
                float ex = __expf(v - vmax);
                s1 += ex; s2 += v * ex;
            }
            float ent = (vmax + __logf(s1)) - s2 / s1;
            int w = w0 + lm;
            out[NPIX + (size_t)h * WW + w] = ent;      // e1
            inds1_ws[(size_t)h * WW + w]   = imax;     // raw class 0..15
        }
    }
}

// ---------------------------------------------------------------------------
// Stages 2/3: conditional per-pixel GEMV cascade. One wave per pixel;
// 32 lanes = 32 output channels; activation broadcast via v_readlane.
// Weight rows are contiguous across lanes -> fully coalesced; expert
// working sets are L2-resident (73MB stage-2 total; ~2MB/row stage-3).
// ---------------------------------------------------------------------------
template <int STAGE>
__global__ __launch_bounds__(256) void condmul_kernel(
    const float* __restrict__ x_in,
    const float* __restrict__ wA, const float* __restrict__ bA,
    const float* __restrict__ wB, const float* __restrict__ bB,
    const float* __restrict__ wC, const float* __restrict__ bC,
    const int* __restrict__ expert_in,   // stage2: raw inds1; stage3: precomputed expert id
    const int* __restrict__ inds12u_in,  // stage3 only: unclipped inds12
    int* __restrict__ inds12u_out,       // stage2 only
    int* __restrict__ expert3_out,       // stage2 only
    float* __restrict__ out)
{
    const int p = blockIdx.x * 8 + (threadIdx.x >> 5);
    if (p >= NPIX) return;
    const int lane = threadIdx.x & 31;
    const int h = p >> 9;  // WW = 512

    const int cbase = (STAGE == 2) ? 32 : 64;
    const float xl = x_in[(size_t)(cbase + lane) * NPIX + p];

    long e;
    if (STAGE == 2) e = (long)expert_in[p] + 16L * h;
    else            e = (long)expert_in[p];

    // layer A: 32 -> 32, lrelu
    float acc = bA[e * 32 + lane];
    {
        const float* wp = wA + e * 1024;
        #pragma unroll
        for (int i = 0; i < 32; ++i)
            acc += bcast_lane(xl, i) * wp[i * 32 + lane];
    }
    const float x1 = lrelu(acc);

    // layer B: 32 -> 32, lrelu
    acc = bB[e * 32 + lane];
    {
        const float* wp = wB + e * 1024;
        #pragma unroll
        for (int i = 0; i < 32; ++i)
            acc += bcast_lane(x1, i) * wp[i * 32 + lane];
    }
    const float x2 = lrelu(acc);

    // layer C: 32 -> 16 (both half-waves compute identical copies)
    const int col = lane & 15;
    acc = bC[e * 16 + col];
    {
        const float* wp = wC + e * 512;
        #pragma unroll
        for (int i = 0; i < 32; ++i)
            acc += bcast_lane(x2, i) * wp[i * 16 + col];
    }

    // argmax (first-max tie-break) + entropy across 16 logits, width-16 butterflies
    float bm = acc; int bi = col;
    #pragma unroll
    for (int off = 1; off < 16; off <<= 1) {
        float ov = __shfl_xor(bm, off, 16);
        int   oi = __shfl_xor(bi, off, 16);
        if (ov > bm || (ov == bm && oi < bi)) { bm = ov; bi = oi; }
    }
    float ex = __expf(acc - bm);
    float s1 = ex, s2 = acc * ex;
    #pragma unroll
    for (int off = 1; off < 16; off <<= 1) {
        s1 += __shfl_xor(s1, off, 16);
        s2 += __shfl_xor(s2, off, 16);
    }
    const float ent = (bm + __logf(s1)) - s2 / s1;

    if (lane == 0) {
        if (STAGE == 2) {
            int inds1 = expert_in[p];                     // 0..15
            int i12u  = inds1 * NC2 + (bi - PP2);         // unclipped
            int i12c  = min(max(i12u, 0), NC1 * NC2 - 1); // 0..191
            inds12u_out[p] = i12u;
            expert3_out[p] = i12c + (NC1 * NC2) * h;
            out[2 * (size_t)NPIX + p] = ent;              // e2
        } else {
            int i12u = inds12u_in[p];
            int i123 = min(max(i12u * NC3 + (bi - PP3), 0), NC1 * NC2 * NC3 - 1);
            out[p] = (float)i123;                          // inds123
            out[3 * (size_t)NPIX + p] = ent;               // e3
        }
    }
}

// ---------------------------------------------------------------------------
extern "C" void kernel_launch(void* const* d_in, const int* in_sizes, int n_in,
                              void* d_out, int out_size, void* d_ws, size_t ws_size,
                              hipStream_t stream) {
    // setup_inputs() dict order (w/b interleaved per layer):
    const float* x_in = (const float*)d_in[0];
    const float* w1_0 = (const float*)d_in[1];
    const float* b1_0 = (const float*)d_in[2];
    const float* w1_1 = (const float*)d_in[3];
    const float* b1_1 = (const float*)d_in[4];
    const float* w1_2 = (const float*)d_in[5];
    const float* b1_2 = (const float*)d_in[6];
    const float* w2_0 = (const float*)d_in[7];
    const float* b2_0 = (const float*)d_in[8];
    const float* w2_1 = (const float*)d_in[9];
    const float* b2_1 = (const float*)d_in[10];
    const float* w2_2 = (const float*)d_in[11];
    const float* b2_2 = (const float*)d_in[12];
    const float* w3_0 = (const float*)d_in[13];
    const float* b3_0 = (const float*)d_in[14];
    const float* w3_1 = (const float*)d_in[15];
    const float* b3_1 = (const float*)d_in[16];
    const float* w3_2 = (const float*)d_in[17];
    const float* b3_2 = (const float*)d_in[18];

    float* out = (float*)d_out;
    int* ws_inds1 = (int*)d_ws;          // [NPIX] raw stage-1 class
    int* ws_i12u  = ws_inds1 + NPIX;     // [NPIX] unclipped inds12
    int* ws_exp3  = ws_i12u + NPIX;      // [NPIX] stage-3 expert id

    stage1_kernel<<<HH, 256, 0, stream>>>(x_in, w1_0, b1_0, w1_1, b1_1, w1_2, b1_2,
                                          out, ws_inds1);
    condmul_kernel<2><<<NPIX / 8, 256, 0, stream>>>(x_in, w2_0, b2_0, w2_1, b2_1, w2_2, b2_2,
                                                    ws_inds1, nullptr, ws_i12u, ws_exp3, out);
    condmul_kernel<3><<<NPIX / 8, 256, 0, stream>>>(x_in, w3_0, b3_0, w3_1, b3_1, w3_2, b3_2,
                                                    ws_exp3, ws_i12u, nullptr, nullptr, out);
}
